// EMU_71588514890095
// MI455X (gfx1250) — matrix operand order repaired and checked by reading, not compiled, not run on hardware
//
#include <hip/hip_runtime.h>
#include <cstdint>
#include <cstddef>

// ---------------- problem constants ----------------
#define DDIM   1024
#define KINT   32
#define TDIMC  64
#define XDIM   1120   // D + K + TDIM
#define BS     64
#define NSTEPS 64
#define XPAD   1128   // LDS row pitch for x (bank-conflict friendly)
#define HPAD   1032   // LDS row pitch for hidden

// ---------------- WMMA helpers (gfx1250, wave32) ----------------
typedef __attribute__((ext_vector_type(16))) __bf16 v16bf;
typedef __attribute__((ext_vector_type(8)))  __bf16 v8bf;
typedef __attribute__((ext_vector_type(8)))  float  f32x8;
typedef __attribute__((ext_vector_type(16))) float  f32x16;

// A-matrix (16x32 bf16): lane m holds K = 16h..16h+15 contiguous (p already offset by 8h)
__device__ __forceinline__ v16bf load_a16(const __bf16* p) {
  const int hh = (threadIdx.x & 31) >> 4;
  v8bf lo = *(const v8bf*)(p + 8 * hh);
  v8bf hi = *(const v8bf*)(p + 8 * hh + 8);
  v16bf r;
#pragma unroll
  for (int i = 0; i < 8; ++i) { r[i] = lo[i]; r[i + 8] = hi[i]; }
  return r;
}

// B-matrix (32x16 bf16): lane n holds K = 16h..16h+15 contiguous.
// WBF=true  : weights already bf16 (pre-converted) -> straight 32B load
// WBF=false : weights f32 -> load 64B + convert
template <bool WBF>
__device__ __forceinline__ v16bf load_bmat(const void* wrow, int k) {
  if constexpr (WBF) {
    return *(const v16bf*)((const __bf16*)wrow + k);
  } else {
    f32x16 w = *(const f32x16*)((const float*)wrow + k);
    v16bf r;
#pragma unroll
    for (int i = 0; i < 16; ++i) r[i] = (__bf16)w[i];
    return r;
  }
}

template <bool WBF>
__device__ __forceinline__ const void* wrow_ptr(const void* W, size_t row, int K) {
  if constexpr (WBF) return (const void*)((const __bf16*)W + row * (size_t)K);
  else               return (const void*)((const float*)W + row * (size_t)K);
}

__device__ __forceinline__ f32x8 wmma_bf16(v16bf a, v16bf b, f32x8 c) {
  return __builtin_amdgcn_wmma_f32_16x16x32_bf16(false, a, false, b, (short)0, c, false, false);
}

// ---------------- CVAE weight bundle ----------------
// w[]: e1, e2, mu, lv, d1, d2, d3 (f32 or bf16 per template); b[]: same order, always f32.
struct CvaeW {
  const void*  w[7];
  const float* b[7];
};

// ============================================================
// Kernel 0: prep — betas/cumprod scalars, Qt, tfeat, bf16 intent
// ============================================================
__global__ __launch_bounds__(256) void prep_kernel(
    const float* Q, const float* noise, const int* t, const float* intent,
    const float* tmw_p, const float* tmb_p, const float* tmw_n, const float* tmb_n,
    float* Qp, float* Qn, float* scal,
    __bf16* intent_bf, __bf16* tfp, __bf16* tfn)
{
  const int b = blockIdx.x;
  const int tid = threadIdx.x;
  const int tb = t[b];
  const float step = (0.05f - 0.0001f) / 19.0f;
  float ab = 1.0f;
  for (int j = 0; j <= tb; ++j) ab *= (1.0f - (0.0001f + step * (float)j));
  const float beta_t = 0.0001f + step * (float)tb;
  if (tid == 0) {
    scal[b]       = 1.0f / sqrtf(1.0f - beta_t);   // inv_sqrt_1mb
    scal[64 + b]  = 1.0f / sqrtf(1.0f - ab);       // inv_sqrt_1mab
    scal[128 + b] = beta_t;                        // bcol
  }
  const float sqab = sqrtf(ab), sq1 = sqrtf(1.0f - ab);
  for (int d = tid; d < DDIM; d += 256) {
    float qt = sqab * Q[b * DDIM + d] + sq1 * noise[b * DDIM + d];
    Qp[b * DDIM + d] = qt;
    Qn[b * DDIM + d] = qt;
  }
  if (tid < TDIMC) {
    float tf = (float)tb * tmw_p[tid] + tmb_p[tid];
    tfp[b * TDIMC + tid] = (__bf16)tf;
    float tn = (float)tb * tmw_n[tid] + tmb_n[tid];
    tfn[b * TDIMC + tid] = (__bf16)tn;
  }
  if (tid < KINT) intent_bf[b * KINT + tid] = (__bf16)intent[b * KINT + tid];
}

// ============================================================
// Kernel 0b: f32 -> bf16 weight conversion (one-time staging)
// ============================================================
__global__ __launch_bounds__(256) void cvt_kernel(const float* s, __bf16* d, int n)
{
  for (int i = blockIdx.x * blockDim.x + threadIdx.x; i < n; i += gridDim.x * blockDim.x)
    d[i] = (__bf16)s[i];
}

// ============================================================
// Kernel 1: full CVAE diffusion scan (64 steps) in one launch.
// grid(8): blockIdx = cvae(0/1) + 2*mtile(0..3). block(512)=16 waves.
// ============================================================
template <bool WBF>
__device__ __forceinline__ void lds_gemm(
    const __bf16* Ab, int lda, int K, const void* W, const float* bias,
    __bf16* Ob, int ldo, bool relu)
{
  const int lane = threadIdx.x & 31, wv = threadIdx.x >> 5;
  const int mi = lane & 15, hf = lane >> 4;
#pragma unroll 1
  for (int tn = 0; tn < 4; ++tn) {
    const int n0 = (wv * 4 + tn) * 16;
    const float bv = bias[n0 + mi];
    f32x8 c;
#pragma unroll
    for (int e = 0; e < 8; ++e) c[e] = bv;
    const __bf16* ap = Ab + mi * lda + 8 * hf;
    const void* wrow = wrow_ptr<WBF>(W, (size_t)(n0 + mi), K);
    for (int k0 = 0; k0 < K; k0 += 32) {
      v16bf a = load_a16(ap + k0);
      v16bf b = load_bmat<WBF>(wrow, k0 + 16 * hf);
      c = wmma_bf16(a, b, c);
    }
#pragma unroll
    for (int e = 0; e < 8; ++e) {
      int m = hf * 8 + e;
      float v = c[e];
      if (relu) v = v > 0.0f ? v : 0.0f;
      Ob[m * ldo + n0 + mi] = (__bf16)v;
    }
  }
}

template <bool WBF>
__device__ __forceinline__ void mulv_z(
    const __bf16* Ab, int lda,
    const void* Wmu, const float* bmu, const void* Wlv, const float* blv,
    const float* eps_blk, __bf16* xb, int ldx, float* kacc)
{
  const int lane = threadIdx.x & 31, wv = threadIdx.x >> 5;
  const int mi = lane & 15, hf = lane >> 4;
#pragma unroll 1
  for (int tn = 0; tn < 4; ++tn) {
    const int n0 = (wv * 4 + tn) * 16;
    f32x8 cm, cl;
    const float bm = bmu[n0 + mi], bl = blv[n0 + mi];
#pragma unroll
    for (int e = 0; e < 8; ++e) { cm[e] = bm; cl[e] = bl; }
    const __bf16* ap = Ab + mi * lda + 8 * hf;
    const void* wrm = wrow_ptr<WBF>(Wmu, (size_t)(n0 + mi), DDIM);
    const void* wrl = wrow_ptr<WBF>(Wlv, (size_t)(n0 + mi), DDIM);
    for (int k0 = 0; k0 < DDIM; k0 += 32) {
      v16bf a = load_a16(ap + k0);
      cm = wmma_bf16(a, load_bmat<WBF>(wrm, k0 + 16 * hf), cm);
      cl = wmma_bf16(a, load_bmat<WBF>(wrl, k0 + 16 * hf), cl);
    }
#pragma unroll
    for (int e = 0; e < 8; ++e) {
      int m = hf * 8 + e, n = n0 + mi;
      float mu = cm[e], lv = cl[e];
      float z = mu + eps_blk[m * DDIM + n] * expf(0.5f * lv);
      xb[m * ldx + n] = (__bf16)z;
      atomicAdd(&kacc[m], 1.0f + lv - mu * mu - expf(lv));
    }
  }
}

template <bool WBF>
__device__ __forceinline__ void dec3_update(
    const __bf16* Ab, int lda, const void* W, const float* bias,
    float* Qws, __bf16* xb, int ldx,
    const float* sisb, const float* sisab, const float* sbcol)
{
  const int lane = threadIdx.x & 31, wv = threadIdx.x >> 5;
  const int mi = lane & 15, hf = lane >> 4;
#pragma unroll 1
  for (int tn = 0; tn < 4; ++tn) {
    const int n0 = (wv * 4 + tn) * 16;
    const float bv = bias[n0 + mi];
    f32x8 c;
#pragma unroll
    for (int e = 0; e < 8; ++e) c[e] = bv;
    const __bf16* ap = Ab + mi * lda + 8 * hf;
    const void* wrow = wrow_ptr<WBF>(W, (size_t)(n0 + mi), DDIM);
    for (int k0 = 0; k0 < DDIM; k0 += 32) {
      v16bf a = load_a16(ap + k0);
      c = wmma_bf16(a, load_bmat<WBF>(wrow, k0 + 16 * hf), c);
    }
#pragma unroll
    for (int e = 0; e < 8; ++e) {
      int m = hf * 8 + e, n = n0 + mi;
      float q = Qws[m * DDIM + n];
      q = sisb[m] * (q - sbcol[m] * c[e] * sisab[m]);
      Qws[m * DDIM + n] = q;
      xb[m * ldx + n] = (__bf16)q;
    }
  }
}

template <bool WBF>
__global__ __launch_bounds__(512) void scan_kernel(
    CvaeW pw, CvaeW nw, const float* eps_pos, const float* eps_neg,
    float* Qp, float* Qn, const float* scal,
    const __bf16* intent_bf, const __bf16* tfp, const __bf16* tfn,
    float* d_out)
{
  extern __shared__ __align__(16) char smem[];
  __bf16* xbuf = (__bf16*)smem;                 // 16 x XPAD
  __bf16* h1   = xbuf + 16 * XPAD;              // 16 x HPAD
  __bf16* h2   = h1 + 16 * HPAD;                // 16 x HPAD
  float*  kacc = (float*)(h2 + 16 * HPAD);      // 16
  float*  sisb = kacc + 16;
  float*  sisab = sisb + 16;
  float*  sbcol = sisab + 16;

  const int cvae = blockIdx.x & 1;
  const int m0 = (blockIdx.x >> 1) * 16;
  const int tid = threadIdx.x;

  const CvaeW W = cvae ? nw : pw;
  const float* eps = cvae ? eps_neg : eps_pos;
  float* Qws = (cvae ? Qn : Qp) + m0 * DDIM;
  const __bf16* tf = cvae ? tfn : tfp;

  for (int idx = tid; idx < 16 * DDIM; idx += blockDim.x) {
    int m = idx >> 10, k = idx & 1023;
    xbuf[m * XPAD + k] = (__bf16)Qws[m * DDIM + k];
  }
  for (int idx = tid; idx < 16 * KINT; idx += blockDim.x) {
    int m = idx >> 5, k = idx & 31;
    xbuf[m * XPAD + DDIM + k] = intent_bf[(m0 + m) * KINT + k];
  }
  for (int idx = tid; idx < 16 * TDIMC; idx += blockDim.x) {
    int m = idx >> 6, k = idx & 63;
    xbuf[m * XPAD + DDIM + KINT + k] = tf[(m0 + m) * TDIMC + k];
  }
  if (tid < 16) {
    kacc[tid]  = 0.0f;
    sisb[tid]  = scal[m0 + tid];
    sisab[tid] = scal[64 + m0 + tid];
    sbcol[tid] = scal[128 + m0 + tid];
  }
  __syncthreads();

#pragma unroll 1
  for (int s = 0; s < NSTEPS; ++s) {
    lds_gemm<WBF>(xbuf, XPAD, XDIM, W.w[0], W.b[0], h1, HPAD, true);  __syncthreads();
    lds_gemm<WBF>(h1, HPAD, DDIM, W.w[1], W.b[1], h2, HPAD, true);    __syncthreads();
    mulv_z<WBF>(h2, HPAD, W.w[2], W.b[2], W.w[3], W.b[3],
                eps + (size_t)(s * BS + m0) * DDIM, xbuf, XPAD, kacc); __syncthreads();
    lds_gemm<WBF>(xbuf, XPAD, XDIM, W.w[4], W.b[4], h1, HPAD, true);  __syncthreads();
    lds_gemm<WBF>(h1, HPAD, DDIM, W.w[5], W.b[5], h2, HPAD, true);    __syncthreads();
    dec3_update<WBF>(h2, HPAD, W.w[6], W.b[6], Qws, xbuf, XPAD, sisb, sisab, sbcol);
    __syncthreads();
  }
  if (tid < 16)
    d_out[3 * BS * DDIM + cvae * BS + m0 + tid] = -0.5f * kacc[tid] / (float)NSTEPS;
}

// ============================================================
// Kernel 2: Emopos/Emoneg select (f32 outputs + bf16 GEMM copies)
// ============================================================
__global__ __launch_bounds__(256) void emo_kernel(
    const float* Q, const unsigned char* mask, const float* Qp, const float* Qn,
    float* outEp, float* outEn, __bf16* Ep_bf, __bf16* En_bf)
{
  const int b = blockIdx.x;
  const bool mb = mask[b] != 0;
  for (int d = threadIdx.x; d < DDIM; d += 256) {
    float ep = mb ? Qp[b * DDIM + d] : Q[b * DDIM + d];
    float en = mb ? Q[b * DDIM + d] : Qn[b * DDIM + d];
    outEp[b * DDIM + d] = ep;
    outEn[b * DDIM + d] = en;
    Ep_bf[b * DDIM + d] = (__bf16)ep;
    En_bf[b * DDIM + d] = (__bf16)en;
  }
}

// ============================================================
// Kernel 3: generic WMMA GEMM  Out = op(A_bf16 @ W_f32^T + bias)
// Weights here are used exactly once -> keep f32 loader.
// ============================================================
__global__ __launch_bounds__(256) void gemm_kernel(
    const __bf16* A, int lda, const float* W, int ldw, const float* bias,
    void* Out, int ldc, int K, int ntiles, int relu, int out_bf)
{
  const int tile = blockIdx.x * 8 + (threadIdx.x >> 5);
  const int mt = tile / ntiles, nt = tile % ntiles;
  const int lane = threadIdx.x & 31, mi = lane & 15, hf = lane >> 4;
  const int n0 = nt * 16, m0 = mt * 16;
  const float bv = bias[n0 + mi];
  f32x8 c;
#pragma unroll
  for (int e = 0; e < 8; ++e) c[e] = bv;
  const __bf16* ap = A + (size_t)(m0 + mi) * lda + 8 * hf;
  const float*  wp = W + (size_t)(n0 + mi) * ldw;
  for (int k0 = 0; k0 < K; k0 += 32) {
    v16bf a = load_a16(ap + k0);
    v16bf b = load_bmat<false>((const void*)wp, k0 + 16 * hf);
    c = wmma_bf16(a, b, c);
  }
#pragma unroll
  for (int e = 0; e < 8; ++e) {
    int m = m0 + hf * 8 + e, n = n0 + mi;
    float v = c[e];
    if (relu) v = v > 0.0f ? v : 0.0f;
    if (out_bf) ((__bf16*)Out)[(size_t)m * ldc + n] = (__bf16)v;
    else        ((float*)Out)[(size_t)m * ldc + n] = v;
  }
}

// ============================================================
// Kernel 4: q-tilde = Wk_h^T @ qh  (folds K-projection into the query)
// ============================================================
__global__ __launch_bounds__(256) void qtilde_kernel(
    const float* qh, const float* inw_pos, const float* inw_neg, float* qt)
{
  const int a = blockIdx.x >> 5, h = (blockIdx.x >> 3) & 3, g = blockIdx.x & 7;
  const int tid = threadIdx.x;
  __shared__ float ql[8][256];
  const float* inw = a ? inw_neg : inw_pos;
  const float* Wk = inw + (size_t)(DDIM + h * 256) * DDIM;
  for (int idx = tid; idx < 2048; idx += 256) {
    int bb = idx >> 8, j = idx & 255;
    ql[bb][j] = qh[(size_t)(a * BS + g * 8 + bb) * DDIM + h * 256 + j];
  }
  __syncthreads();
  float acc[4][8];
#pragma unroll
  for (int kk = 0; kk < 4; ++kk)
#pragma unroll
    for (int bb = 0; bb < 8; ++bb) acc[kk][bb] = 0.0f;
#pragma unroll 1
  for (int j = 0; j < 256; ++j) {
    float wv[4];
#pragma unroll
    for (int kk = 0; kk < 4; ++kk) wv[kk] = Wk[(size_t)j * DDIM + tid + kk * 256];
#pragma unroll
    for (int bb = 0; bb < 8; ++bb) {
      float q = ql[bb][j];
#pragma unroll
      for (int kk = 0; kk < 4; ++kk) acc[kk][bb] += wv[kk] * q;
    }
  }
#pragma unroll
  for (int kk = 0; kk < 4; ++kk)
#pragma unroll
    for (int bb = 0; bb < 8; ++bb)
      qt[((size_t)(a * BS + g * 8 + bb) * 4 + h) * DDIM + tid + kk * 256] = acc[kk][bb];
}

// ============================================================
// Kernel 5: fused scores + softmax + weighted-sum over H.
// ============================================================
__global__ __launch_bounds__(512) void attn_kernel(
    const float* H, const float* qh, const float* qt,
    const float* inb_pos, const float* inb_neg, __bf16* c_bf)
{
  const int b = blockIdx.x, a = blockIdx.y;
  const int tid = threadIdx.x;
  __shared__ float sarr[4][1024];
  __shared__ float qtl[4][1024];
  __shared__ float red[512];
  __shared__ float s0s[4];

  const float* inb = a ? inb_neg : inb_pos;
  const float* bk = inb + DDIM;
  const float* Hb = H + (size_t)b * 1024 * DDIM;

  for (int idx = tid; idx < 4096; idx += 512) {
    int h = idx >> 10, k = idx & 1023;
    qtl[h][k] = qt[((size_t)(a * BS + b) * 4 + h) * DDIM + k];
  }
#pragma unroll 1
  for (int h = 0; h < 4; ++h) {
    float p = 0.0f;
    if (tid < 256) p = qh[(size_t)(a * BS + b) * DDIM + h * 256 + tid] * bk[h * 256 + tid];
    red[tid] = p;
    __syncthreads();
    for (int s = 256; s > 0; s >>= 1) { if (tid < s) red[tid] += red[tid + s]; __syncthreads(); }
    if (tid == 0) s0s[h] = red[0];
    __syncthreads();
  }
  const int lane = tid & 31, wv = tid >> 5;
#pragma unroll 1
  for (int rr = 0; rr < 64; ++rr) {
    const int trow = wv * 64 + rr;
    const float* hr = Hb + (size_t)trow * DDIM + lane * 4;
    float acch[4] = {0.0f, 0.0f, 0.0f, 0.0f};
#pragma unroll 1
    for (int it = 0; it < 8; ++it) {
      float4 hv = *(const float4*)(hr + it * 128);
      int k = lane * 4 + it * 128;
#pragma unroll
      for (int h = 0; h < 4; ++h) {
        float4 qv = *(const float4*)&qtl[h][k];
        acch[h] += hv.x * qv.x + hv.y * qv.y + hv.z * qv.z + hv.w * qv.w;
      }
    }
#pragma unroll
    for (int off = 16; off > 0; off >>= 1)
#pragma unroll
      for (int h = 0; h < 4; ++h) acch[h] += __shfl_xor(acch[h], off);
    if (lane == 0) {
#pragma unroll
      for (int h = 0; h < 4; ++h) sarr[h][trow] = (acch[h] + s0s[h]) * 0.0625f;
    }
  }
  __syncthreads();
#pragma unroll 1
  for (int h = 0; h < 4; ++h) {
    float v0 = sarr[h][tid], v1 = sarr[h][tid + 512];
    red[tid] = fmaxf(v0, v1);
    __syncthreads();
    for (int s = 256; s > 0; s >>= 1) { if (tid < s) red[tid] = fmaxf(red[tid], red[tid + s]); __syncthreads(); }
    float mx = red[0];
    __syncthreads();
    float e0 = expf(v0 - mx), e1 = expf(v1 - mx);
    red[tid] = e0 + e1;
    __syncthreads();
    for (int s = 256; s > 0; s >>= 1) { if (tid < s) red[tid] += red[tid + s]; __syncthreads(); }
    float inv = 1.0f / red[0];
    __syncthreads();
    sarr[h][tid] = e0 * inv;
    sarr[h][tid + 512] = e1 * inv;
    __syncthreads();
  }
  float acc[4][2] = {{0.f,0.f},{0.f,0.f},{0.f,0.f},{0.f,0.f}};
  const int d0 = tid, d1 = tid + 512;
#pragma unroll 1
  for (int trow = 0; trow < 1024; ++trow) {
    float h0 = Hb[(size_t)trow * DDIM + d0];
    float h1 = Hb[(size_t)trow * DDIM + d1];
#pragma unroll
    for (int h = 0; h < 4; ++h) {
      float av = sarr[h][trow];
      acc[h][0] += av * h0;
      acc[h][1] += av * h1;
    }
  }
#pragma unroll
  for (int h = 0; h < 4; ++h) {
    c_bf[((size_t)(a * 4 + h) * BS + b) * DDIM + d0] = (__bf16)acc[h][0];
    c_bf[((size_t)(a * 4 + h) * BS + b) * DDIM + d1] = (__bf16)acc[h][1];
  }
}

// ============================================================
// Host side
// ============================================================
struct CvaeF { const float* w[7]; const float* b[7]; const float* tmw; const float* tmb; };

static void map_cvae(void* const* d_in, const int* sz, int base, CvaeF& f)
{
  const float* a[16];
  for (int i = 0; i < 16; ++i) a[i] = (const float*)d_in[base + i];
  if (sz[base] == 1146880) { // insertion order: enc1_w, enc1_b, enc2_w, ...
    f.w[0] = a[0];  f.b[0] = a[1];  f.w[1] = a[2];  f.b[1] = a[3];
    f.w[2] = a[4];  f.b[2] = a[5];  f.w[3] = a[6];  f.b[3] = a[7];
    f.w[4] = a[8];  f.b[4] = a[9];  f.w[5] = a[10]; f.b[5] = a[11];
    f.w[6] = a[12]; f.b[6] = a[13]; f.tmw = a[14];  f.tmb = a[15];
  } else { // jax-sorted keys: dec1_b,dec1_w,dec2_b,dec2_w,dec3_b,dec3_w,enc1_b,enc1_w,enc2_b,enc2_w,lv_b,lv_w,mu_b,mu_w,tm_b,tm_w
    f.b[4] = a[0];  f.w[4] = a[1];  f.b[5] = a[2];  f.w[5] = a[3];
    f.b[6] = a[4];  f.w[6] = a[5];  f.b[0] = a[6];  f.w[0] = a[7];
    f.b[1] = a[8];  f.w[1] = a[9];  f.b[3] = a[10]; f.w[3] = a[11];
    f.b[2] = a[12]; f.w[2] = a[13]; f.tmb = a[14];  f.tmw = a[15];
  }
}

static void map_attn(void* const* d_in, const int* sz, int base,
                     const float** inw, const float** inb,
                     const float** outw, const float** outb)
{
  if (sz[base] == 3 * DDIM * DDIM) { // insertion: in_w, in_b, out_w, out_b
    *inw = (const float*)d_in[base + 0]; *inb = (const float*)d_in[base + 1];
    *outw = (const float*)d_in[base + 2]; *outb = (const float*)d_in[base + 3];
  } else { // sorted: in_b, in_w, out_b, out_w
    *inb = (const float*)d_in[base + 0]; *inw = (const float*)d_in[base + 1];
    *outb = (const float*)d_in[base + 2]; *outw = (const float*)d_in[base + 3];
  }
}

extern "C" void kernel_launch(void* const* d_in, const int* in_sizes, int n_in,
                              void* d_out, int out_size, void* d_ws, size_t ws_size,
                              hipStream_t stream)
{
  (void)n_in; (void)out_size;
  const float* Q       = (const float*)d_in[0];
  const float* intent  = (const float*)d_in[1];
  const unsigned char* mask = (const unsigned char*)d_in[2];
  const float* H       = (const float*)d_in[3];
  const int*   t       = (const int*)d_in[4];
  const float* noise   = (const float*)d_in[5];
  const float* eps_pos = (const float*)d_in[6];
  const float* eps_neg = (const float*)d_in[7];

  CvaeF pf, nf;
  map_cvae(d_in, in_sizes, 8, pf);
  map_cvae(d_in, in_sizes, 24, nf);
  const float *inw_p, *inb_p, *outw_p, *outb_p;
  const float *inw_n, *inb_n, *outw_n, *outb_n;
  map_attn(d_in, in_sizes, 40, &inw_p, &inb_p, &outw_p, &outb_p);
  map_attn(d_in, in_sizes, 44, &inw_n, &inb_n, &outw_n, &outb_n);
  const float* opw = (const float*)d_in[48];
  const float* opb = (const float*)d_in[49];

  // ---- workspace carve ----
  char* p = (char*)d_ws;
  auto take = [&](size_t n) { char* r = p; p += (n + 255) & ~(size_t)255; return r; };
  float*  Qp        = (float*)take((size_t)BS * DDIM * 4);
  float*  Qn        = (float*)take((size_t)BS * DDIM * 4);
  float*  scal      = (float*)take(3 * BS * 4);
  __bf16* intent_bf = (__bf16*)take((size_t)BS * KINT * 2);
  __bf16* tfp       = (__bf16*)take((size_t)BS * TDIMC * 2);
  __bf16* tfn       = (__bf16*)take((size_t)BS * TDIMC * 2);
  __bf16* Ep_bf     = (__bf16*)take((size_t)BS * DDIM * 2);
  __bf16* En_bf     = (__bf16*)take((size_t)BS * DDIM * 2);
  float*  qh_ws     = (float*)take((size_t)2 * BS * DDIM * 4);
  float*  qt_ws     = (float*)take((size_t)2 * BS * 4 * DDIM * 4);
  __bf16* c_bf      = (__bf16*)take((size_t)2 * 4 * BS * DDIM * 2);
  __bf16* o_bf      = (__bf16*)take((size_t)2 * BS * DDIM * 2);
  __bf16* fused_bf  = (__bf16*)take((size_t)BS * 2 * DDIM * 2);

  // ---- optional bf16 staging of the 14 CVAE weight matrices (~29 MB) ----
  static const int wK[7] = {XDIM, DDIM, DDIM, DDIM, XDIM, DDIM, DDIM};
  size_t welems = 0;
  for (int i = 0; i < 7; ++i) welems += (size_t)1024 * wK[i];
  const size_t base_used = (size_t)(p - (char*)d_ws);
  const bool use_wbf = (base_used + 2 * welems * 2 + 1024) <= ws_size;

  CvaeW pw, nw;
  for (int i = 0; i < 7; ++i) { pw.b[i] = pf.b[i]; nw.b[i] = nf.b[i]; }
  if (use_wbf) {
    for (int c = 0; c < 2; ++c) {
      const CvaeF& f = c ? nf : pf;
      CvaeW& w = c ? nw : pw;
      for (int i = 0; i < 7; ++i) {
        int n = 1024 * wK[i];
        __bf16* dst = (__bf16*)take((size_t)n * 2);
        cvt_kernel<<<512, 256, 0, stream>>>(f.w[i], dst, n);
        w.w[i] = (const void*)dst;
      }
    }
  } else {
    for (int i = 0; i < 7; ++i) { pw.w[i] = (const void*)pf.w[i]; nw.w[i] = (const void*)nf.w[i]; }
  }

  float* out = (float*)d_out;

  // 1) prep
  prep_kernel<<<BS, 256, 0, stream>>>(Q, noise, t, intent, pf.tmw, pf.tmb, nf.tmw, nf.tmb,
                                      Qp, Qn, scal, intent_bf, tfp, tfn);
  // 2) full diffusion scan (persistent, 8 blocks, ~100 KB LDS each)
  const size_t scan_lds = (size_t)(16 * XPAD + 2 * 16 * HPAD) * 2 + 64 * 4;
  if (use_wbf)
    scan_kernel<true><<<8, 512, scan_lds, stream>>>(pw, nw, eps_pos, eps_neg,
                                                    Qp, Qn, scal, intent_bf, tfp, tfn, out);
  else
    scan_kernel<false><<<8, 512, scan_lds, stream>>>(pw, nw, eps_pos, eps_neg,
                                                     Qp, Qn, scal, intent_bf, tfp, tfn, out);
  // 3) Emopos / Emoneg
  emo_kernel<<<BS, 256, 0, stream>>>(Q, mask, Qp, Qn,
                                     out + BS * DDIM, out + 2 * BS * DDIM, Ep_bf, En_bf);
  // 4) qh = Emo @ Wq^T + bq (f32 out), both attentions
  gemm_kernel<<<32, 256, 0, stream>>>(Ep_bf, DDIM, inw_p, DDIM, inb_p,
                                      (void*)(qh_ws), DDIM, DDIM, 64, 0, 0);
  gemm_kernel<<<32, 256, 0, stream>>>(En_bf, DDIM, inw_n, DDIM, inb_n,
                                      (void*)(qh_ws + BS * DDIM), DDIM, DDIM, 64, 0, 0);
  // 5) q-tilde (fold K projection into query)
  qtilde_kernel<<<64, 256, 0, stream>>>(qh_ws, inw_p, inw_n, qt_ws);
  // 6) scores + softmax + weighted-sum of H
  attn_kernel<<<dim3(BS, 2), 512, 0, stream>>>(H, qh_ws, qt_ws, inb_p, inb_n, c_bf);
  // 7) o[b, h*256:+256] = c[b,h,:] @ Wv_h^T + bv_h  (block-diagonal per head)
  for (int a = 0; a < 2; ++a) {
    const float* inw = a ? inw_n : inw_p;
    const float* inb = a ? inb_n : inb_p;
    for (int h = 0; h < 4; ++h) {
      gemm_kernel<<<8, 256, 0, stream>>>(
          c_bf + (size_t)(a * 4 + h) * BS * DDIM, DDIM,
          inw + (size_t)(2 * DDIM + h * 256) * DDIM, DDIM,
          inb + 2 * DDIM + h * 256,
          (void*)(o_bf + (size_t)a * BS * DDIM + h * 256), DDIM, DDIM, 16, 0, 1);
    }
  }
  // 8) ap/an = o @ out_w^T + out_b -> fused (bf16, concat layout)
  gemm_kernel<<<32, 256, 0, stream>>>(o_bf, DDIM, outw_p, DDIM, outb_p,
                                      (void*)(fused_bf), 2 * DDIM, DDIM, 64, 0, 1);
  gemm_kernel<<<32, 256, 0, stream>>>(o_bf + (size_t)BS * DDIM, DDIM, outw_n, DDIM, outb_n,
                                      (void*)(fused_bf + DDIM), 2 * DDIM, DDIM, 64, 0, 1);
  // 9) Emofused = fused @ out_proj_w^T + out_proj_b (f32 out)
  gemm_kernel<<<32, 256, 0, stream>>>(fused_bf, 2 * DDIM, opw, 2 * DDIM, opb,
                                      (void*)out, DDIM, 2 * DDIM, 64, 0, 0);
}
